// Encoder_40982577938507
// MI455X (gfx1250) — matrix operand order repaired and checked by reading, not compiled, not run on hardware
//
#include <hip/hip_runtime.h>

// ---------------------------------------------------------------------------
// Transformer encoder layer for MI455X (gfx1250, wave32, WMMA).
// - All matmuls: v_wmma_f32_16x16x32_bf16 (bf16 operands, f32 accumulate).
// - Weights are converted/transposed to bf16 [N][K] once, so GEMM staging is
//   pure contiguous 16B copies -> GLOBAL_LOAD_ASYNC_TO_LDS_B128 double-buffer
//   (guarded by __has_builtin; synchronous fallback otherwise).
// - Reference softmax is over the QUERY axis; the key-only additive mask is a
//   per-column constant and cancels exactly -> skipped. Softmax needs
//   per-COLUMN (per-key) max/sum stats: pass A computes them, pass B applies.
// - V is produced transposed ([B,H,DK,S]) so attention pass B needs no LDS
//   staging for V and runs with zero workgroup barriers.
// ---------------------------------------------------------------------------

typedef __attribute__((ext_vector_type(16))) __bf16 v16bf;
typedef __attribute__((ext_vector_type(8)))  float  v8f;
typedef __attribute__((ext_vector_type(4)))  int    v4i;

#define S_LEN   1024
#define D_MODEL 1024
#define NHEAD   16
#define DK_     64
#define DFF_    4096
#define BATCH   8

#if defined(__gfx1250__) && \
    __has_builtin(__builtin_amdgcn_global_load_async_to_lds_b128) && \
    __has_builtin(__builtin_amdgcn_s_wait_asynccnt)
#define USE_ASYNC_COPY 1
#else
#define USE_ASYNC_COPY 0
#endif

#if USE_ASYNC_COPY
// Builtin signature: (v4i addrspace(1)* src, v4i addrspace(3)* dst, Imm, Imm).
typedef __attribute__((address_space(1))) v4i* g128p;
typedef __attribute__((address_space(3))) v4i* l128p;
#endif

__device__ __forceinline__ void cp_b128(const __bf16* g, __bf16* l) {
#if USE_ASYNC_COPY
  // ptr -> integer -> AS-qualified ptr: always-legal casts; on AMDGCN the
  // generic global bit-pattern equals AS1, and generic LDS addr[31:0] is the
  // AS3 offset (ISA aperture rules), so the values are correct.
  __builtin_amdgcn_global_load_async_to_lds_b128(
      (g128p)(unsigned long long)(g),
      (l128p)(unsigned int)(unsigned long long)(l), 0, 0);
#else
  *(uint4*)l = *(const uint4*)g;
#endif
}

#if USE_ASYNC_COPY
#define WAIT_ASYNC(n) __builtin_amdgcn_s_wait_asynccnt(n)
#else
#define WAIT_ASYNC(n) ((void)0)
#endif

__device__ __forceinline__ v8f wmma_bf16(v16bf a, v16bf b, v8f c) {
  return __builtin_amdgcn_wmma_f32_16x16x32_bf16(
      /*neg_a=*/false, a, /*neg_b=*/false, b,
      /*c_mod=*/(short)0, c, /*reuse_a=*/false, /*reuse_b=*/false);
}

// A-operand fragment (16 rows x 32 K), rows contiguous with stride `ld` bf16.
__device__ __forceinline__ v16bf frag_a16x32(const __bf16* base, int ld,
                                             int l15, int half) {
  const __bf16* p = base + l15 * ld + 8 * half;
  union { v16bf v; uint4 q[2]; } u;
  u.q[0] = *(const uint4*)(p);
  u.q[1] = *(const uint4*)(p + 16);
  return u.v;
}

// B-operand fragment (32 K x 16 cols) from a [col][K] (transposed) layout.
__device__ __forceinline__ v16bf frag_b32x16(const __bf16* base, int ld,
                                             int l15, int half) {
  const __bf16* p = base + l15 * ld + 8 * half;
  union { v16bf v; uint4 q[2]; } u;
  u.q[0] = *(const uint4*)(p);
  u.q[1] = *(const uint4*)(p + 16);
  return u.v;
}

// ---------------------------------------------------------------------------
// Batched GEMM: C[z] = A_bf16[z] (MxK row-major) * Bt_bf16[z]^T (Bt is [N][K])
//               + bias, block tile 128(M) x 64(N), K-step 32, double-buffered
//               async LDS staging. 8 waves; wave w owns rows w*16..w*16+15.
// MODE 0: f32 (+bias). MODE 1: bf16 (+bias). MODE 2: bf16 relu(+bias).
// MODE 3: bf16 (+bias), stored TRANSPOSED: C[n][m] with leading dim ldc.
// ---------------------------------------------------------------------------
template <int MODE>
__global__ void __launch_bounds__(256) gemm_bf16_wmma(
    const __bf16* __restrict__ A, int lda, long aStride, int aDiv,
    const __bf16* __restrict__ Bt, int ldbt, long bStride, int bMod,
    const float*  __restrict__ bias, int biasStride,
    void* __restrict__ Cout, int ldc, long cStride,
    int K)
{
  __shared__ __bf16 As[2][128][40];   // 80B padded rows: conflict-free b128
  __shared__ __bf16 Bs[2][64][40];
  const int z    = blockIdx.z;
  const int t    = threadIdx.x;
  const int wave = t >> 5, lane = t & 31;
  const int l15  = lane & 15, half = lane >> 4;
  const int n0   = blockIdx.x * 64;
  const int m0   = blockIdx.y * 128;

  A  += (long)(z / aDiv) * aStride;
  Bt += (long)(z % bMod) * bStride;
  const float* bi = bias + (long)(z % bMod) * biasStride + n0;

  v8f acc[4] = {};

  // Staging decomposition: per stage each thread issues exactly 3 b128 copies
  // (2 for the 128x32 A tile, 1 for the 64x32 B tile) -> wave-uniform ASYNCcnt.
  const int rowA = t >> 1, colA = (t & 1) * 16;
  const int rowB = t >> 2, colB = (t & 3) * 8;

  auto stage = [&](int buf, int k0) {
    const __bf16* sa = A + (long)(m0 + rowA) * lda + k0 + colA;
    cp_b128(sa,     &As[buf][rowA][colA]);
    cp_b128(sa + 8, &As[buf][rowA][colA + 8]);
    cp_b128(Bt + (long)(n0 + rowB) * ldbt + k0 + colB, &Bs[buf][rowB][colB]);
  };

  stage(0, 0);
  const int nsteps = K >> 5;
  for (int s = 0; s < nsteps; ++s) {
    const int buf = s & 1;
    if (s + 1 < nsteps) {
      stage(buf ^ 1, (s + 1) * 32);  // prefetch next tile
      WAIT_ASYNC(3);                 // current tile's 3 copies retired
    } else {
      WAIT_ASYNC(0);
    }
    __syncthreads();
    v16bf af = frag_a16x32(&As[buf][wave * 16][0], 40, l15, half);
    #pragma unroll
    for (int c = 0; c < 4; ++c) {
      v16bf bfr = frag_b32x16(&Bs[buf][c * 16][0], 40, l15, half);
      acc[c] = wmma_bf16(af, bfr, acc[c]);
    }
    __syncthreads();
  }

  // Epilogue. C fragment: VGPR r, lane -> (row = r + 8*half, col = l15).
  const int rowBase = m0 + wave * 16 + 8 * half;
  #pragma unroll
  for (int c = 0; c < 4; ++c) {
    int col  = c * 16 + l15;
    float bv = bi[col];
    #pragma unroll
    for (int r = 0; r < 8; ++r) {
      float v = acc[c][r] + bv;
      if (MODE == 3) {
        long idx = (long)z * cStride + (long)(n0 + col) * ldc + rowBase + r;
        ((__bf16*)Cout)[idx] = (__bf16)v;
      } else {
        long idx = (long)z * cStride + (long)(rowBase + r) * ldc + n0 + col;
        if (MODE == 0)      ((float*)Cout)[idx]  = v;
        else if (MODE == 1) ((__bf16*)Cout)[idx] = (__bf16)v;
        else                ((__bf16*)Cout)[idx] = (__bf16)fmaxf(v, 0.0f);
      }
    }
  }
}

// ---------------------------------------------------------------------------
// Attention pass A: per-column (per-key) online max/sum of exp over queries.
// Wave w owns 16 columns, scans all 1024 query rows with WMMA score tiles.
// C layout puts 8 rows of ONE column in each lane -> in-lane reduction, then
// shfl_xor(16) merges the two lane halves.
// ---------------------------------------------------------------------------
__global__ void __launch_bounds__(256) attn_colstats(
    const __bf16* __restrict__ qb, const __bf16* __restrict__ kb,
    float* __restrict__ colM, float* __restrict__ colZ)
{
  const int bh  = blockIdx.y;
  const int t   = threadIdx.x;
  const int wave = t >> 5, lane = t & 31;
  const int l15 = lane & 15, half = lane >> 4;
  const int j0  = blockIdx.x * 128 + wave * 16;

  const __bf16* kbase = kb + ((long)bh * S_LEN + j0) * DK_;
  v16bf kf0 = frag_b32x16(kbase,      DK_, l15, half);   // K = 0..31
  v16bf kf1 = frag_b32x16(kbase + 32, DK_, l15, half);   // K = 32..63
  const __bf16* qbh = qb + (long)bh * S_LEN * DK_;

  float m = -3.0e38f, zsum = 0.0f;
  for (int i0 = 0; i0 < S_LEN; i0 += 16) {
    v16bf a0 = frag_a16x32(qbh + (long)i0 * DK_,      DK_, l15, half);
    v16bf a1 = frag_a16x32(qbh + (long)i0 * DK_ + 32, DK_, l15, half);
    v8f s = {};
    s = wmma_bf16(a0, kf0, s);
    s = wmma_bf16(a1, kf1, s);
    float tm = m;
    #pragma unroll
    for (int r = 0; r < 8; ++r) tm = fmaxf(tm, s[r] * 0.125f);
    float rescale = __expf(m - tm);
    float part = 0.0f;
    #pragma unroll
    for (int r = 0; r < 8; ++r) part += __expf(s[r] * 0.125f - tm);
    zsum = zsum * rescale + part;
    m = tm;
  }
  float mo = __shfl_xor(m, 16, 32);
  float zo = __shfl_xor(zsum, 16, 32);
  float mc = fmaxf(m, mo);
  float zc = zsum * __expf(m - mc) + zo * __expf(mo - mc);
  if (half == 0) {
    colM[(long)bh * S_LEN + j0 + l15] = mc;
    colZ[(long)bh * S_LEN + j0 + l15] = zc;
  }
}

// ---------------------------------------------------------------------------
// Attention pass B: ctx[i,:] = sum_j exp(s[i,j]-M_j)/Z_j * v[j,:].
// V arrives TRANSPOSED ([B,H,DK,S]) -> V B-fragments load straight from
// global; the only LDS use is the wave-private P C-layout -> A-layout
// round-trip, so the kernel needs no workgroup barriers at all.
// ctx written bf16 in [B,S,H*DK] order (A operand for the Wo GEMM).
// ---------------------------------------------------------------------------
__global__ void __launch_bounds__(256) attn_ctx(
    const __bf16* __restrict__ qb, const __bf16* __restrict__ kb,
    const __bf16* __restrict__ vt,
    const float* __restrict__ colM, const float* __restrict__ colZ,
    __bf16* __restrict__ ctx)
{
  __shared__ __bf16 Pt[128][40];   // per-wave 16x32 P tiles
  const int bh = blockIdx.y;
  const int b  = bh >> 4, h = bh & 15;
  const int t  = threadIdx.x;
  const int wave = t >> 5, lane = t & 31;
  const int l15 = lane & 15, half = lane >> 4;
  const int i0  = blockIdx.x * 128 + wave * 16;

  const __bf16* qrow = qb + ((long)bh * S_LEN + i0) * DK_;
  v16bf aq0 = frag_a16x32(qrow,      DK_, l15, half);
  v16bf aq1 = frag_a16x32(qrow + 32, DK_, l15, half);
  const __bf16* vbh = vt + (long)bh * DK_ * S_LEN;

  v8f acc[4] = {};

  for (int j0 = 0; j0 < S_LEN; j0 += 32) {
    #pragma unroll
    for (int jt = 0; jt < 2; ++jt) {
      const __bf16* kbase = kb + ((long)bh * S_LEN + j0 + jt * 16) * DK_;
      v16bf k0 = frag_b32x16(kbase,      DK_, l15, half);
      v16bf k1 = frag_b32x16(kbase + 32, DK_, l15, half);
      v8f s = {};
      s = wmma_bf16(aq0, k0, s);
      s = wmma_bf16(aq1, k1, s);
      int col  = j0 + jt * 16 + l15;
      float Mj = colM[(long)bh * S_LEN + col];
      float rZ = 1.0f / colZ[(long)bh * S_LEN + col];
      #pragma unroll
      for (int r = 0; r < 8; ++r) {
        float p = __expf(s[r] * 0.125f - Mj) * rZ;
        Pt[wave * 16 + r + 8 * half][jt * 16 + l15] = (__bf16)p;
      }
    }
    // Same-wave LDS write->read: DS ops are in-order within a wave.
    v16bf ap = frag_a16x32(&Pt[wave * 16][0], 40, l15, half);  // K=j 0..31
    #pragma unroll
    for (int c = 0; c < 4; ++c) {
      v16bf vf = frag_b32x16(vbh + (long)(c * 16) * S_LEN + j0, S_LEN,
                             l15, half);
      acc[c] = wmma_bf16(ap, vf, acc[c]);
    }
  }

  const long rowbase = ((long)b * S_LEN + i0 + 8 * half) * D_MODEL + h * DK_;
  #pragma unroll
  for (int c = 0; c < 4; ++c)
    #pragma unroll
    for (int r = 0; r < 8; ++r)
      ctx[rowbase + (long)r * D_MODEL + c * 16 + l15] = (__bf16)acc[c][r];
}

// ---------------------------------------------------------------------------
// Residual + LayerNorm over D=1024. One 256-thread block per row.
// ---------------------------------------------------------------------------
__global__ void __launch_bounds__(256) ln_residual(
    const float* __restrict__ in1, const float* __restrict__ in2,
    const float* __restrict__ g, const float* __restrict__ bvec,
    float* __restrict__ outF, __bf16* __restrict__ outB)
{
  __shared__ float red[256];
  const int t = threadIdx.x;
  const long row = blockIdx.x;
  const float* a = in1 + row * D_MODEL;
  const float* c = in2 + row * D_MODEL;
  float loc[4]; float s = 0.0f;
  #pragma unroll
  for (int i = 0; i < 4; ++i) {
    int col = t + i * 256;
    loc[i] = a[col] + c[col];
    s += loc[i];
  }
  red[t] = s; __syncthreads();
  for (int o = 128; o > 0; o >>= 1) { if (t < o) red[t] += red[t + o]; __syncthreads(); }
  float mu = red[0] * (1.0f / D_MODEL);
  __syncthreads();
  float vs = 0.0f;
  #pragma unroll
  for (int i = 0; i < 4; ++i) { float d = loc[i] - mu; vs += d * d; }
  red[t] = vs; __syncthreads();
  for (int o = 128; o > 0; o >>= 1) { if (t < o) red[t] += red[t + o]; __syncthreads(); }
  float inv = rsqrtf(red[0] * (1.0f / D_MODEL) + 1e-5f);
  #pragma unroll
  for (int i = 0; i < 4; ++i) {
    int col = t + i * 256;
    float yv = (loc[i] - mu) * inv * g[col] + bvec[col];
    outF[row * D_MODEL + col] = yv;
    if (outB) outB[row * D_MODEL + col] = (__bf16)yv;
  }
}

__global__ void __launch_bounds__(256) cvt_f32_bf16(
    const float* __restrict__ in, __bf16* __restrict__ out)
{
  long i = ((long)blockIdx.x * 256 + threadIdx.x) * 4;
  float4 v = *(const float4*)(in + i);
  out[i + 0] = (__bf16)v.x;
  out[i + 1] = (__bf16)v.y;
  out[i + 2] = (__bf16)v.z;
  out[i + 3] = (__bf16)v.w;
}

// One-time weight convert+transpose: in f32 [K][N] -> out bf16 [N][K].
// Grid (N/64, K/64, batch). Coalesced reads over n; one-shot cost.
__global__ void __launch_bounds__(256) cvt_transpose_bf16(
    const float* __restrict__ in, __bf16* __restrict__ out, int K, int N)
{
  const int z = blockIdx.z;
  in  += (long)z * K * N;
  out += (long)z * K * N;
  const int t = threadIdx.x;
  const int n = blockIdx.x * 64 + (t & 63);
  const int kbase = blockIdx.y * 64 + (t >> 6);
  #pragma unroll
  for (int i = 0; i < 16; ++i) {
    int k = kbase + i * 4;
    out[(long)n * K + k] = (__bf16)in[(long)k * N + n];
  }
}

// ---------------------------------------------------------------------------
extern "C" void kernel_launch(void* const* d_in, const int* in_sizes, int n_in,
                              void* d_out, int out_size, void* d_ws, size_t ws_size,
                              hipStream_t stream) {
  (void)in_sizes; (void)n_in; (void)out_size; (void)ws_size;
  const float* x  = (const float*)d_in[0];
  // d_in[1] = binary_mask: varies only over keys; reference softmax is over
  // the query axis, so the additive mask cancels exactly -> unused.
  const float* Wq = (const float*)d_in[2];
  const float* bq = (const float*)d_in[3];
  const float* Wk = (const float*)d_in[4];
  const float* bk = (const float*)d_in[5];
  const float* Wv = (const float*)d_in[6];
  const float* bv = (const float*)d_in[7];
  const float* Wo = (const float*)d_in[8];
  const float* bo = (const float*)d_in[9];
  const float* ga = (const float*)d_in[10];
  const float* ba = (const float*)d_in[11];
  const float* W1 = (const float*)d_in[12];
  const float* b1 = (const float*)d_in[13];
  const float* W2 = (const float*)d_in[14];
  const float* b2 = (const float*)d_in[15];
  const float* gf = (const float*)d_in[16];
  const float* bfv = (const float*)d_in[17];

  char* ws = (char*)d_ws;
  size_t off = 0;
  auto alloc = [&](size_t bytes) -> void* {
    void* p = ws + off;
    off = (off + bytes + 255) & ~(size_t)255;
    return p;
  };
  const long MROWS = (long)BATCH * S_LEN;                  // 8192
  __bf16* xb   = (__bf16*)alloc(MROWS * D_MODEL * 2);      // x in bf16
  __bf16* qb   = (__bf16*)alloc(MROWS * D_MODEL * 2);      // [B,H,S,DK]
  __bf16* kbuf = (__bf16*)alloc(MROWS * D_MODEL * 2);      // [B,H,S,DK]
  __bf16* vtb  = (__bf16*)alloc(MROWS * D_MODEL * 2);      // [B,H,DK,S]
  float*  colM = (float*)alloc((long)BATCH * NHEAD * S_LEN * 4);
  float*  colZ = (float*)alloc((long)BATCH * NHEAD * S_LEN * 4);
  __bf16* ctxb = (__bf16*)alloc(MROWS * D_MODEL * 2);      // [B,S,D]
  float*  attn = (float*)alloc(MROWS * D_MODEL * 4);
  float*  yf   = (float*)alloc(MROWS * D_MODEL * 4);
  __bf16* yb   = (__bf16*)alloc(MROWS * D_MODEL * 2);
  __bf16* hb   = (__bf16*)alloc(MROWS * DFF_ * 2);
  float*  f2   = (float*)alloc(MROWS * D_MODEL * 4);
  // transposed bf16 weights
  __bf16* WqT = (__bf16*)alloc((long)NHEAD * D_MODEL * DK_ * 2);  // [H][DK][D]
  __bf16* WkT = (__bf16*)alloc((long)NHEAD * D_MODEL * DK_ * 2);
  __bf16* WvT = (__bf16*)alloc((long)NHEAD * D_MODEL * DK_ * 2);
  __bf16* WoT = (__bf16*)alloc((long)D_MODEL * D_MODEL * 2);      // [D][D]
  __bf16* W1T = (__bf16*)alloc((long)D_MODEL * DFF_ * 2);         // [DFF][D]
  __bf16* W2T = (__bf16*)alloc((long)D_MODEL * DFF_ * 2);         // [D][DFF]

  dim3 blk(256);

  // 0) activations -> bf16; weights -> bf16 transposed [N][K] (one-time).
  cvt_f32_bf16<<<dim3((unsigned)(MROWS * D_MODEL / 1024)), blk, 0, stream>>>(x, xb);
  cvt_transpose_bf16<<<dim3(1, 16, NHEAD), blk, 0, stream>>>(Wq, WqT, D_MODEL, DK_);
  cvt_transpose_bf16<<<dim3(1, 16, NHEAD), blk, 0, stream>>>(Wk, WkT, D_MODEL, DK_);
  cvt_transpose_bf16<<<dim3(1, 16, NHEAD), blk, 0, stream>>>(Wv, WvT, D_MODEL, DK_);
  cvt_transpose_bf16<<<dim3(16, 16, 1), blk, 0, stream>>>(Wo, WoT, D_MODEL, D_MODEL);
  cvt_transpose_bf16<<<dim3(64, 16, 1), blk, 0, stream>>>(W1, W1T, D_MODEL, DFF_);
  cvt_transpose_bf16<<<dim3(16, 64, 1), blk, 0, stream>>>(W2, W2T, DFF_, D_MODEL);

  // 1) QKV projections, batched over B*H. V is stored transposed (MODE 3).
  dim3 gQKV(1, S_LEN / 128, BATCH * NHEAD);
  gemm_bf16_wmma<1><<<gQKV, blk, 0, stream>>>(
      xb, D_MODEL, (long)S_LEN * D_MODEL, NHEAD,
      WqT, D_MODEL, (long)DK_ * D_MODEL, NHEAD, bq, DK_,
      qb, DK_, (long)S_LEN * DK_, D_MODEL);
  gemm_bf16_wmma<1><<<gQKV, blk, 0, stream>>>(
      xb, D_MODEL, (long)S_LEN * D_MODEL, NHEAD,
      WkT, D_MODEL, (long)DK_ * D_MODEL, NHEAD, bk, DK_,
      kbuf, DK_, (long)S_LEN * DK_, D_MODEL);
  gemm_bf16_wmma<3><<<gQKV, blk, 0, stream>>>(
      xb, D_MODEL, (long)S_LEN * D_MODEL, NHEAD,
      WvT, D_MODEL, (long)DK_ * D_MODEL, NHEAD, bv, DK_,
      vtb, S_LEN, (long)DK_ * S_LEN, D_MODEL);

  // 2) attention: column softmax stats, then barrier-free context.
  dim3 gS(S_LEN / 128, BATCH * NHEAD);
  attn_colstats<<<gS, blk, 0, stream>>>(qb, kbuf, colM, colZ);
  attn_ctx<<<gS, blk, 0, stream>>>(qb, kbuf, vtb, colM, colZ, ctxb);

  // 3) output projection: attn = ctx @ Wo + bo (f32).
  dim3 gO(D_MODEL / 64, (unsigned)(MROWS / 128), 1);
  gemm_bf16_wmma<0><<<gO, blk, 0, stream>>>(
      ctxb, D_MODEL, 0, 1, WoT, D_MODEL, 0, 1, bo, 0,
      attn, D_MODEL, 0, D_MODEL);

  // 4) y = LN(x + attn); also emit bf16 copy.
  ln_residual<<<dim3((unsigned)MROWS), blk, 0, stream>>>(x, attn, ga, ba, yf, yb);

  // 5) h = relu(y @ W1 + b1) -> bf16.
  dim3 g1(DFF_ / 64, (unsigned)(MROWS / 128), 1);
  gemm_bf16_wmma<2><<<g1, blk, 0, stream>>>(
      yb, D_MODEL, 0, 1, W1T, D_MODEL, 0, 1, b1, 0,
      hb, DFF_, 0, D_MODEL);

  // 6) f2 = h @ W2 + b2 (f32).
  dim3 g2(D_MODEL / 64, (unsigned)(MROWS / 128), 1);
  gemm_bf16_wmma<0><<<g2, blk, 0, stream>>>(
      hb, DFF_, 0, 1, W2T, DFF_, 0, 1, b2, 0,
      f2, D_MODEL, 0, DFF_);

  // 7) out = LN(y + f2).
  ln_residual<<<dim3((unsigned)MROWS), blk, 0, stream>>>(
      yf, f2, gf, bfv, (float*)d_out, (__bf16*)nullptr);
}
